// CFConv_3204045603903
// MI455X (gfx1250) — matrix operand order repaired and checked by reading, not compiled, not run on hardware
//
#include <hip/hip_runtime.h>

// ---------------- CDNA5 (gfx1250) CFConv as implicit WMMA-GEMM ----------------
// out[b,o,y,x] = sum_{k,i,j} weight[k,o,i,j] * fa[b,i,y+dy,x+dx] * wa[b,j,y+dy,x+dx]
// Main GEMM:  M=o(64), N=pixel, K=(k,j,i<64) = 9792  (306 chunks of 32, f16 WMMA)
// Bias GEMM:  K=(k,j) = 153 (pad 160) for the fa-ones channel (i==64).
// wa_j (incl. ones j==16) folded into B fragments as a per-column splat scale.
// Each wave: M=16 x N=32 tile, 4 independent accumulator chains for WMMA ILP.

typedef __attribute__((ext_vector_type(16))) _Float16 v16h;
typedef __attribute__((ext_vector_type(8)))  _Float16 v8h;
typedef __attribute__((ext_vector_type(8)))  float    v8f;

#define BATCH 4
#define WCH   16
#define FCH   64
#define OCH   64
#define H_    128
#define W_    128
#define HO    126
#define WO    126
#define KJ    153                 // 9*17 (k,j) pairs
#define NC1   (KJ * 2)            // 306 main K-chunks of 32
#define NC2   5                   // bias K-chunks (153 padded to 160)
#define A1_ELEMS (NC1 * OCH * 32) // 626688 halfs
#define A2_ELEMS (NC2 * OCH * 32) // 10240 halfs
#define CHUNK_STRIDE (OCH * 32)   // halfs between consecutive K-chunks in A1

#define TILE_X    64              // output columns per block
#define HALO_X    (TILE_X + 2)    // 66
#define NPOS      (3 * HALO_X)    // 198 staged pixel positions
#define FA_STRIDE 72              // 64 ch + 8 pad (halfs) -> 144B rows, 16B aligned
#define WA_STRIDE 18              // 17 + 1 pad

__device__ __forceinline__ v16h splat16(_Float16 x) {
  v16h r;
#pragma unroll
  for (int t = 0; t < 16; ++t) r[t] = x;
  return r;
}

#define SHUF16(a, b) __builtin_shufflevector((a), (b), 0,1,2,3,4,5,6,7,8,9,10,11,12,13,14,15)

__device__ __forceinline__ v16h loadB(const _Float16* p) {
  v8h lo = *(const v8h*)(p);
  v8h hi = *(const v8h*)(p + 8);
  return SHUF16(lo, hi);
}

__device__ __forceinline__ v16h loadA(const _Float16* row, int hi8) {
  v8h lo = *(const v8h*)(row + (hi8 << 1));       // K {0..7} or {16..23}
  v8h hi = *(const v8h*)(row + (hi8 << 1) + 8);   // K {8..15} or {24..31}
  return SHUF16(lo, hi);
}

// ---- pack weight (f32) into f16 A-fragment rows, K-major per chunk ----
__global__ void cfconv_prepack(const float* __restrict__ w,
                               _Float16* __restrict__ a1,
                               _Float16* __restrict__ a2) {
  int idx = blockIdx.x * blockDim.x + threadIdx.x;
  if (idx < A1_ELEMS) {
    int t  = idx & 31;          // K-local within chunk
    int o  = (idx >> 5) & 63;   // output channel (M row)
    int c  = idx >> 11;         // chunk id
    int kj = c >> 1;
    int k  = kj / 17, j = kj - k * 17;
    int i  = ((c & 1) << 5) + t;                       // i in [0,64)
    a1[idx] = (_Float16)w[((k * OCH + o) * 65 + i) * 17 + j];
  } else if (idx < A1_ELEMS + A2_ELEMS) {
    int id2 = idx - A1_ELEMS;
    int t   = id2 & 31;
    int o   = (id2 >> 5) & 63;
    int c2  = id2 >> 11;
    int kjt = (c2 << 5) + t;
    float v = 0.f;
    if (kjt < KJ) {
      int k = kjt / 17, j = kjt - k * 17;
      v = w[((k * OCH + o) * 65 + 64) * 17 + j];       // i == 64 (fa ones row)
    }
    a2[id2] = (_Float16)v;
  }
}

__global__ __launch_bounds__(256)
void cfconv_main(const float* __restrict__ inw, const float* __restrict__ inf,
                 const _Float16* __restrict__ a1, const _Float16* __restrict__ a2,
                 float* __restrict__ out) {
  __shared__ __align__(16) _Float16 faLDS[NPOS * FA_STRIDE];   // 28.5 KB
  __shared__ __align__(16) _Float16 waLDS[NPOS * WA_STRIDE];   //  7.1 KB

  const int tid = threadIdx.x;
  const int x0  = blockIdx.x * TILE_X;   // 0 or 64
  const int y   = blockIdx.y;            // output row (< 126)
  const int b   = blockIdx.z;

  // ---- cooperative stage: 3 x 66 halo of inputf (f32 -> f16) ----
  for (int idx = tid; idx < FCH * NPOS; idx += 256) {
    int ch = idx / NPOS;
    int r  = idx - ch * NPOS;
    int ry = r / HALO_X, rx = r - ry * HALO_X;
    int xg = x0 + rx;
    float v = 0.f;
    if (xg < W_) v = inf[(((size_t)b * FCH + ch) * H_ + (y + ry)) * W_ + xg];
    faLDS[(ry * HALO_X + rx) * FA_STRIDE + ch] = (_Float16)v;
  }
  // ---- 3 x 66 halo of inputw, ones channel at j==16 ----
  for (int idx = tid; idx < NPOS * 17; idx += 256) {
    int pos = idx / 17;
    int j   = idx - pos * 17;
    int ry  = pos / HALO_X, rx = pos - ry * HALO_X;
    int xg  = x0 + rx;
    float v;
    if (j == 16)      v = 1.f;
    else if (xg < W_) v = inw[(((size_t)b * WCH + j) * H_ + (y + ry)) * W_ + xg];
    else              v = 0.f;
    waLDS[pos * WA_STRIDE + j] = (_Float16)v;
  }
  __syncthreads();

  // ---- wave tiling: 8 waves = 4 M-tiles x 2 N-groups of 32 columns ----
  const int wave  = tid >> 5;
  const int lane  = tid & 31;
  const int hi    = lane >> 4;       // half-wave selector (ISA K/M split)
  const int hi8   = hi << 3;
  const int ln    = lane & 15;       // fragment row / column index
  const int Mbase = (wave >> 1) << 4;
  const int xs    = (wave & 1) << 5; // 0 or 32 within the 64-col tile

  v8f acc00 = {}, acc01 = {}, acc10 = {}, acc11 = {};

  // A row base for this lane (chunk 0), advanced by CHUNK_STRIDE per chunk
  const _Float16* aptr = a1 + ((size_t)(Mbase + ln) << 5);

  // ---- main GEMM: 9 shifts x 17 j x (2 chunks of 32 i) ----
  for (int dy = 0; dy < 3; ++dy) {
    for (int dx = 0; dx < 3; ++dx) {
      const int pos0 = dy * HALO_X + xs + dx + ln;   // column group 0
      const int pos1 = pos0 + 16;                    // column group 1
      const _Float16* frow0 = &faLDS[pos0 * FA_STRIDE];
      const _Float16* frow1 = &faLDS[pos1 * FA_STRIDE];
      const _Float16* wrow0 = &waLDS[pos0 * WA_STRIDE];
      const _Float16* wrow1 = &waLDS[pos1 * WA_STRIDE];
      for (int j = 0; j < 17; ++j) {
        v16h ws0 = splat16(wrow0[j]);
        v16h ws1 = splat16(wrow1[j]);
        // two K-chunks (i halves) -> two independent A fragments
        v16h av0 = loadA(aptr, hi8);                aptr += CHUNK_STRIDE;
        v16h av1 = loadA(aptr, hi8);                aptr += CHUNK_STRIDE;
        // B fragments: K-consecutive 16 halfs per half-wave, scaled by wa_j
        v16h b00 = loadB(frow0 + (hi << 4))      * ws0;  // cols 0, i 0..31
        v16h b01 = loadB(frow0 + 32 + (hi << 4)) * ws0;  // cols 0, i 32..63
        v16h b10 = loadB(frow1 + (hi << 4))      * ws1;  // cols 1, i 0..31
        v16h b11 = loadB(frow1 + 32 + (hi << 4)) * ws1;  // cols 1, i 32..63
        // 4 independent accumulator chains
        acc00 = __builtin_amdgcn_wmma_f32_16x16x32_f16(false, av0, false, b00,
                                                       (short)0, acc00, false, false);
        acc01 = __builtin_amdgcn_wmma_f32_16x16x32_f16(false, av1, false, b01,
                                                       (short)0, acc01, false, false);
        acc10 = __builtin_amdgcn_wmma_f32_16x16x32_f16(false, av0, false, b10,
                                                       (short)0, acc10, false, false);
        acc11 = __builtin_amdgcn_wmma_f32_16x16x32_f16(false, av1, false, b11,
                                                       (short)0, acc11, false, false);
      }
    }
  }

  // ---- bias GEMM for fa ones channel: K = 153 (k,j), padded to 160 ----
#pragma unroll
  for (int c2 = 0; c2 < NC2; ++c2) {
    const _Float16* arow = a2 + ((size_t)(c2 * OCH + Mbase + ln) << 5);
    v16h av = loadA(arow, hi8);
    v16h bv0, bv1;
#pragma unroll
    for (int t = 0; t < 16; ++t) {
      int kjt = (c2 << 5) + (hi << 4) + t;
      _Float16 v0 = (_Float16)0.f, v1 = (_Float16)0.f;
      if (kjt < KJ) {
        int k  = kjt / 17, j = kjt - k * 17;
        int dy = k / 3,   dx = k - dy * 3;
        int pos = dy * HALO_X + xs + dx + ln;
        v0 = waLDS[pos * WA_STRIDE + j];
        v1 = waLDS[(pos + 16) * WA_STRIDE + j];
      }
      bv0[t] = v0;
      bv1[t] = v1;
    }
    acc00 = __builtin_amdgcn_wmma_f32_16x16x32_f16(false, av, false, bv0,
                                                   (short)0, acc00, false, false);
    acc10 = __builtin_amdgcn_wmma_f32_16x16x32_f16(false, av, false, bv1,
                                                   (short)0, acc10, false, false);
  }

  // ---- combine split-K chains ----
  v8f accN0 = acc00 + acc01;
  v8f accN1 = acc10 + acc11;

  // ---- epilogue: C/D layout VGPR r -> M = r (lanes 0-15) / 8+r (lanes 16-31) ----
  const int xA = x0 + xs + ln;
  const int xB = xA + 16;
#pragma unroll
  for (int r2 = 0; r2 < 8; ++r2) {
    const int o = Mbase + (hi << 3) + r2;
    const size_t rowbase = (((size_t)b * OCH + o) * HO + y) * WO;
    if (xA < WO) out[rowbase + xA] = accN0[r2];
    if (xB < WO) out[rowbase + xB] = accN1[r2];
  }
}

extern "C" void kernel_launch(void* const* d_in, const int* in_sizes, int n_in,
                              void* d_out, int out_size, void* d_ws, size_t ws_size,
                              hipStream_t stream) {
  (void)in_sizes; (void)n_in; (void)out_size; (void)ws_size;
  const float* inw = (const float*)d_in[0];   // inputw (4,16,128,128)
  const float* inf = (const float*)d_in[1];   // inputf (4,64,128,128)
  const float* wgt = (const float*)d_in[2];   // weight (9,64,65,17)
  _Float16* a1 = (_Float16*)d_ws;             // 1.22 MB packed main weights
  _Float16* a2 = a1 + A1_ELEMS;               // 20 KB packed bias weights

  const int ntot = A1_ELEMS + A2_ELEMS;
  cfconv_prepack<<<(ntot + 255) / 256, 256, 0, stream>>>(wgt, a1, a2);
  cfconv_main<<<dim3((WO + TILE_X - 1) / TILE_X, HO, BATCH), 256, 0, stream>>>(
      inw, inf, a1, a2, (float*)d_out);
}
